// GQAS2Attention_74208444940983
// MI455X (gfx1250) — matrix operand order repaired and checked by reading, not compiled, not run on hardware
//
#include <hip/hip_runtime.h>
#include <hip/hip_bf16.h>

#define DEVI __device__ __forceinline__

typedef __attribute__((ext_vector_type(16))) _Float16 v16h;
typedef __attribute__((ext_vector_type(8)))  _Float16 v8h;
typedef __attribute__((ext_vector_type(8)))  float    v8f;
typedef __attribute__((ext_vector_type(4)))  unsigned u32x4;
typedef __attribute__((ext_vector_type(8)))  int      i32x8;
typedef __attribute__((ext_vector_type(4)))  int      i32x4;

// ---- problem constants (from reference) ----
constexpr int B_    = 2;
constexpr int S_    = 2048;
constexpr int HID_  = 1024;
constexpr int H_    = 16;
constexpr int KVH_  = 2;
constexpr int NOPE_ = 192;
constexpr int ROPE_ = 64;
constexpr int D_    = 256;
constexpr int QD_   = H_ * D_;          // 4096
constexpr int GD_   = H_;               // 16
constexpr int KVD_  = KVH_ * D_;        // 512
constexpr int QKVD_ = QD_ + GD_ + KVD_; // 4624
constexpr int BS_   = B_ * S_;          // 4096
constexpr float EPS_ = 1e-6f;

// padded B-row count for the QKV GEMM (grid of 128-wide N tiles)
constexpr int QKV_NBLK_ = (QKVD_ + 127) / 128;   // 37
constexpr int QKV_NPAD_ = QKV_NBLK_ * 128;       // 4736

// ---------------------------------------------------------------------------
// WMMA helpers (CDNA5 wave32, v_wmma_f32_16x16x32_f16)
// ---------------------------------------------------------------------------
DEVI v8f wmma_f16(v16h a, v16h b, v8f c) {
  return __builtin_amdgcn_wmma_f32_16x16x32_f16(
      /*neg_a=*/false, a, /*neg_b=*/false, b,
      /*c_mod=*/(short)0, c, /*reuse_a=*/false, /*reuse_b=*/false);
}

// Build a 16-half fragment from two 8-half (16B) chunks.
// A-fragment (16x32 f16, lane M=lane%16, g=lane/16):
//   halfs 0..7  -> K = 8g + h        (chunk at p)
//   halfs 8..15 -> K = 16 + 8g + h-8 (chunk at p+16)
// B-fragment (32x16 f16, lane N=lane%16, g=lane/16):
//   same K mapping as A-fragment     (chunks at p and p+16)
DEVI v16h make_frag(const _Float16* p0, const _Float16* p1) {
  v8h lo = *(const v8h*)p0;
  v8h hi = *(const v8h*)p1;
  v16h r;
#pragma unroll
  for (int i = 0; i < 8; ++i) { r[i] = lo[i]; r[i + 8] = hi[i]; }
  return r;
}
DEVI v16h afrag(const _Float16* p) { return make_frag(p, p + 16); }
DEVI v16h bfrag(const _Float16* p) { return make_frag(p, p + 16); }

// ---------------------------------------------------------------------------
// Tensor Data Mover: async 2D tile load (global f16 -> LDS), TENSORcnt-tracked.
// D#: count=1, type=2, data_size=1 (2B), 2D tensor, no pad/iterate/multicast.
// lds_byte = workgroup-relative LDS byte offset (low 32 bits of generic ptr).
// ---------------------------------------------------------------------------
DEVI void tdm_load_tile_2d(unsigned lds_byte, const _Float16* gtile,
                           int tile_w, int tile_h, int pitch_elems,
                           int tensor_h) {
  const unsigned long long ga = (unsigned long long)(uintptr_t)gtile;
  u32x4 g0;
  g0[0] = 1u;                                   // count=1, user mode
  g0[1] = lds_byte;                             // lds_addr (bytes)
  g0[2] = (unsigned)ga;                         // global_addr[31:0]
  g0[3] = (unsigned)(ga >> 32) | (2u << 30);    // global_addr[56:32] | type=2
  i32x8 g1;
  g1[0] = (int)(1u << 16);                      // wg_mask=0, data_size=1 (2B)
  g1[1] = (int)(((unsigned)pitch_elems & 0xffffu) << 16);        // dim0 lo16
  g1[2] = (int)((((unsigned)pitch_elems >> 16) & 0xffffu) |
                (((unsigned)tensor_h & 0xffffu) << 16));         // dim0 hi / dim1 lo
  g1[3] = (int)((((unsigned)tensor_h >> 16) & 0xffffu) |
                (((unsigned)tile_w & 0xffffu) << 16));           // dim1 hi / tile_dim0
  g1[4] = (int)((unsigned)tile_h & 0xffffu);                     // tile_dim1, tile_dim2=0
  g1[5] = (int)(unsigned)pitch_elems;                            // dim0_stride lo32
  g1[6] = 0;                                                     // dim0_stride hi / dim1_stride lo
  g1[7] = 0;                                                     // dim1_stride hi
  i32x4 gz = {0, 0, 0, 0};
#if __clang_major__ >= 23
  i32x8 gz8 = {0, 0, 0, 0, 0, 0, 0, 0};
  __builtin_amdgcn_tensor_load_to_lds(g0, g1, gz, gz, gz8, 0);
#else
  __builtin_amdgcn_tensor_load_to_lds(g0, g1, gz, gz, 0);
#endif
}

DEVI unsigned lds_offset_of(const void* p) {
  return (unsigned)(uintptr_t)p;  // generic LDS addr: low 32 bits = LDS offset
}

// ---------------------------------------------------------------------------
// fp32 -> f16 conversion
// ---------------------------------------------------------------------------
__global__ void cvt_f32_f16(const float* __restrict__ in,
                            _Float16* __restrict__ out, int n) {
  int i = blockIdx.x * blockDim.x + threadIdx.x;
  if (i < n) out[i] = (_Float16)in[i];
}

// ---------------------------------------------------------------------------
// NT GEMM with TDM-staged A tiles:
//   C[M,N] (f32) = A[M,K] (f16 row-major) * B[Npad,K]^T (f16 row-major)
// Block = 4 waves, tile M=64 x N=128; wave tile 64x32 (8 WMMA accumulators:
// B frag reused 4x, A frag 2x). Wave 0 DMAs the shared 64x64 A slab into LDS
// (double-buffered, s_wait_tensorcnt + barrier); B read direct (L2-resident).
// Requires M%64==0, K%64==0. Stores guarded by col<N (B rows padded to grid).
// ---------------------------------------------------------------------------
__global__ __launch_bounds__(128) void gemm_nt_f16f32(
    const _Float16* __restrict__ A, const _Float16* __restrict__ Bm,
    float* __restrict__ C, int M, int N, int K) {
  __shared__ _Float16 abuf[2][64 * 64];  // [buf][row*64 + k] : 8KB each
  const int lane = threadIdx.x & 31;
  const int wv   = threadIdx.x >> 5;
  const int g    = lane >> 4;
  const int ln16 = lane & 15;
  const int m0   = blockIdx.y * 64;
  const int n0   = blockIdx.x * 128 + wv * 32;

  v8f acc[8];
#pragma unroll
  for (int i = 0; i < 8; ++i) acc[i] = (v8f){};

  const _Float16* brow0 = Bm + (size_t)(n0 + ln16) * K + 8 * g;
  const _Float16* brow1 = brow0 + (size_t)16 * K;
  const _Float16* Abase = A + (size_t)m0 * K;

  const int steps = K >> 6;  // 64-deep K slabs
  if (wv == 0) {
    tdm_load_tile_2d(lds_offset_of(&abuf[0][0]), Abase, 64, 64, K, M);
    __builtin_amdgcn_s_wait_tensorcnt(0);
  }
  __syncthreads();

  for (int st = 0; st < steps; ++st) {
    const int cur = st & 1;
    if (wv == 0 && st + 1 < steps) {
      tdm_load_tile_2d(lds_offset_of(&abuf[cur ^ 1][0]),
                       Abase + ((st + 1) << 6), 64, 64, K, M);
    }
    const _Float16* ab = &abuf[cur][0];
#pragma unroll
    for (int kc = 0; kc < 64; kc += 32) {
      const int k0 = (st << 6) + kc;
      if (kc == 0 && st + 1 < steps) {  // cache hint for next B slab
        __builtin_prefetch(brow0 + k0 + 64, 0, 3);
        __builtin_prefetch(brow1 + k0 + 64, 0, 3);
      }
      v16h bf0 = bfrag(brow0 + k0);
      v16h bf1 = bfrag(brow1 + k0);
#pragma unroll
      for (int ms = 0; ms < 4; ++ms) {
        const _Float16* ap = ab + (ms * 16 + ln16) * 64 + kc + 8 * g;
        v16h a = make_frag(ap, ap + 16);
        acc[ms]     = wmma_f16(a, bf0, acc[ms]);
        acc[ms + 4] = wmma_f16(a, bf1, acc[ms + 4]);
      }
    }
    if (wv == 0 && st + 1 < steps) __builtin_amdgcn_s_wait_tensorcnt(0);
    __syncthreads();  // next slab visible; current buffer free for overwrite
  }

#pragma unroll
  for (int ns = 0; ns < 2; ++ns) {
    const int col = n0 + ns * 16 + ln16;
    if (col < N) {
#pragma unroll
      for (int ms = 0; ms < 4; ++ms) {
#pragma unroll
        for (int r = 0; r < 8; ++r) {
          C[(size_t)(m0 + ms * 16 + r + 8 * g) * N + col] = acc[ms + ns * 4][r];
        }
      }
    }
  }
}

// ---------------------------------------------------------------------------
// Prep: RMS norm + RoPE for q/k, RoPE + lambda blend for v, sigmoid gate.
// One 256-thread block (8 waves) per token. Each wave: 2 q heads; waves 0,1
// additionally handle the 2 kv heads. Lane owns 8 consecutive dims; the RoPE
// partner (dim +/- 32) lives in lane^4 -> __shfl_xor.
// ---------------------------------------------------------------------------
DEVI float rope_angle(int s, int i /*0..31*/) {
  float inv_freq = __powf(10000.0f, -(float)i * (1.0f / 32.0f));
  return (float)s * inv_freq;
}

__global__ __launch_bounds__(256) void prep_kernel(
    const float* __restrict__ qkv, const float* __restrict__ v_first,
    const float* __restrict__ qnw, const float* __restrict__ knw,
    const float* __restrict__ vlam,
    _Float16* __restrict__ qh,   // [B,H,S,D]
    _Float16* __restrict__ kh,   // [B,KVH,S,D]
    _Float16* __restrict__ vT,   // [B,KVH,D,S]
    float* __restrict__ gate) {  // [BS,H] (sigmoid applied)
  const int t = blockIdx.x;
  const int b = t / S_, s = t % S_;
  const int lane = threadIdx.x & 31;
  const int wv   = threadIdx.x >> 5;
  const float* row = qkv + (size_t)t * QKVD_;

  for (int hh = wv * 2; hh < wv * 2 + 2; ++hh) {
    const float* x = row + hh * D_ + lane * 8;
    float xv[8];
    float ss = 0.f;
#pragma unroll
    for (int e = 0; e < 8; ++e) { xv[e] = x[e]; ss += xv[e] * xv[e]; }
#pragma unroll
    for (int m = 1; m < 32; m <<= 1) ss += __shfl_xor(ss, m, 32);
    float inv = rsqrtf(ss * (1.0f / D_) + EPS_);
    float xn[8];
#pragma unroll
    for (int e = 0; e < 8; ++e) {
      int d = lane * 8 + e;
      xn[e] = xv[e] * inv * qnw[d];
    }
#pragma unroll
    for (int e = 0; e < 8; ++e) {
      int d = lane * 8 + e;
      float partner = __shfl_xor(xn[e], 4, 32);
      float outv = xn[e];
      if (d >= NOPE_) {
        int r = d - NOPE_;
        float ang = rope_angle(s, r & 31);
        float c = cosf(ang), sn = sinf(ang);
        outv = (r < 32) ? (xn[e] * c - partner * sn)
                        : (xn[e] * c + partner * sn);
      }
      qh[((size_t)(b * H_ + hh) * S_ + s) * D_ + d] = (_Float16)outv;
    }
  }

  if (wv < KVH_) {
    const int kvh = wv;
    const float* x = row + QD_ + GD_ + kvh * D_ + lane * 8;
    float xv[8];
    float ss = 0.f;
#pragma unroll
    for (int e = 0; e < 8; ++e) { xv[e] = x[e]; ss += xv[e] * xv[e]; }
#pragma unroll
    for (int m = 1; m < 32; m <<= 1) ss += __shfl_xor(ss, m, 32);
    float inv = rsqrtf(ss * (1.0f / D_) + EPS_);
    float lam0 = 1.f / (1.f + __expf(-vlam[0]));
    float lam1 = 1.f / (1.f + __expf(-vlam[1]));
#pragma unroll
    for (int e = 0; e < 8; ++e) {
      int d = lane * 8 + e;
      float kn = xv[e] * inv * knw[d];
      float kpart = __shfl_xor(kn, 4, 32);
      float vpart = __shfl_xor(xv[e], 4, 32);
      float kout = kn, vout = xv[e];
      if (d >= NOPE_) {
        int r = d - NOPE_;
        float ang = rope_angle(s, r & 31);
        float c = cosf(ang), sn = sinf(ang);
        if (r < 32) { kout = kn * c - kpart * sn; vout = xv[e] * c - vpart * sn; }
        else        { kout = kn * c + kpart * sn; vout = xv[e] * c + vpart * sn; }
      }
      kh[((size_t)(b * KVH_ + kvh) * S_ + s) * D_ + d] = (_Float16)kout;
      float vf = v_first[((size_t)(b * S_ + s) * KVH_ + kvh) * D_ + d];
      float vb = lam0 * vout + lam1 * vf;
      vT[((size_t)(b * KVH_ + kvh) * D_ + d) * S_ + s] = (_Float16)vb;
    }
  }

  if (threadIdx.x < GD_) {
    float gl = row[QD_ + threadIdx.x];
    gate[(size_t)t * GD_ + threadIdx.x] = 1.f / (1.f + __expf(-gl));
  }
}

// ---------------------------------------------------------------------------
// Causal flash attention. One wave per (b, h, 16 query rows); 32 keys/iter.
// Scores via WMMA, online softmax with 16-lane butterfly reductions, P relaid
// C-layout->A-layout through private LDS scratch (s_wait_dscnt fences, no
// block barrier since causal trip counts differ per wave), P x V via WMMA
// with transposed V so B-frags are contiguous.
// ---------------------------------------------------------------------------
__global__ __launch_bounds__(128) void attn_kernel(
    const _Float16* __restrict__ qh,  // [B,H,S,D]
    const _Float16* __restrict__ kh,  // [B,KVH,S,D]
    const _Float16* __restrict__ vT,  // [B,KVH,D,S]
    float* __restrict__ O) {          // [B,S,H,D]
  __shared__ _Float16 pbuf[4][16][33];
  const int lane = threadIdx.x & 31;
  const int wv   = threadIdx.x >> 5;
  const int g    = lane >> 4;
  const int ln16 = lane & 15;
  const int h = blockIdx.y, b = blockIdx.z;
  const int m0  = (blockIdx.x * 4 + wv) * 16;
  const int kvh = h / (H_ / KVH_);

  const _Float16* Qbase = qh + (size_t)(b * H_ + h) * S_ * D_;
  const _Float16* Kbase = kh + (size_t)(b * KVH_ + kvh) * S_ * D_;
  const _Float16* Vbase = vT + (size_t)(b * KVH_ + kvh) * D_ * S_;

  v8f o[16];
#pragma unroll
  for (int tt = 0; tt < 16; ++tt) o[tt] = (v8f){};
  float mrow[8], lrow[8];
#pragma unroll
  for (int r = 0; r < 8; ++r) { mrow[r] = -3.0e38f; lrow[r] = 0.f; }

  const float scale = 0.0625f;  // D^-0.5
  const _Float16* qrow = Qbase + (size_t)(m0 + ln16) * D_ + 8 * g;

  const int nkb = (m0 + 16 + 31) >> 5;
  for (int jb = 0; jb < nkb; ++jb) {
    const int kb0 = jb * 32;
    v8f s0 = {}, s1 = {};
    const _Float16* krow0 = Kbase + (size_t)(kb0 + ln16) * D_ + 8 * g;
    const _Float16* krow1 = krow0 + (size_t)16 * D_;
#pragma unroll
    for (int dc = 0; dc < 8; ++dc) {
      v16h a = afrag(qrow + dc * 32);
      s0 = wmma_f16(a, bfrag(krow0 + dc * 32), s0);
      s1 = wmma_f16(a, bfrag(krow1 + dc * 32), s1);
    }
#pragma unroll
    for (int r = 0; r < 8; ++r) {
      const int q   = m0 + r + 8 * g;
      const int k0i = kb0 + ln16;
      float v0 = s0[r] * scale; if (k0i      > q) v0 = -3.0e38f;
      float v1 = s1[r] * scale; if (k0i + 16 > q) v1 = -3.0e38f;
      float mx = fmaxf(v0, v1);
#pragma unroll
      for (int mm = 1; mm < 16; mm <<= 1) mx = fmaxf(mx, __shfl_xor(mx, mm, 32));
      float mnew = fmaxf(mrow[r], mx);
      float p0 = __expf(v0 - mnew);
      float p1 = __expf(v1 - mnew);
      float psum = p0 + p1;
#pragma unroll
      for (int mm = 1; mm < 16; mm <<= 1) psum += __shfl_xor(psum, mm, 32);
      float alpha = __expf(mrow[r] - mnew);
      mrow[r] = mnew;
      lrow[r] = lrow[r] * alpha + psum;
#pragma unroll
      for (int tt = 0; tt < 16; ++tt) o[tt][r] *= alpha;
      pbuf[wv][r + 8 * g][ln16]      = (_Float16)p0;
      pbuf[wv][r + 8 * g][ln16 + 16] = (_Float16)p1;
    }
    asm volatile("s_wait_dscnt 0" ::: "memory");
    v16h pa;
#pragma unroll
    for (int hh = 0; hh < 16; ++hh) {
      const int kk = (hh < 8) ? (8 * g + hh) : (16 + 8 * g + (hh - 8));
      pa[hh] = pbuf[wv][ln16][kk];
    }
#pragma unroll
    for (int tt = 0; tt < 16; ++tt) {
      const _Float16* vrow = Vbase + (size_t)(tt * 16 + ln16) * S_ + kb0 + 8 * g;
      o[tt] = wmma_f16(pa, bfrag(vrow), o[tt]);
    }
    asm volatile("s_wait_dscnt 0" ::: "memory");
  }

#pragma unroll
  for (int r = 0; r < 8; ++r) {
    const float invl = (lrow[r] > 0.f) ? (1.f / lrow[r]) : 0.f;
    const int q = m0 + r + 8 * g;
#pragma unroll
    for (int tt = 0; tt < 16; ++tt) {
      O[((size_t)(b * S_ + q) * H_ + h) * D_ + tt * 16 + ln16] = o[tt][r] * invl;
    }
  }
}

// ---------------------------------------------------------------------------
// Post: inverse RoPE -> RMS(post_norm_w) -> * sigmoid(gate) -> f16 [BS, QD]
// ---------------------------------------------------------------------------
__global__ __launch_bounds__(256) void post_kernel(
    const float* __restrict__ O, const float* __restrict__ gate,
    const float* __restrict__ pnw, _Float16* __restrict__ att) {
  const int t = blockIdx.x;
  const int s = t % S_;
  const int lane = threadIdx.x & 31;
  const int wv   = threadIdx.x >> 5;
  for (int hh = wv * 2; hh < wv * 2 + 2; ++hh) {
    const float* x = O + ((size_t)t * H_ + hh) * D_ + lane * 8;
    float xv[8], y[8];
#pragma unroll
    for (int e = 0; e < 8; ++e) xv[e] = x[e];
    float ss = 0.f;
#pragma unroll
    for (int e = 0; e < 8; ++e) {
      int d = lane * 8 + e;
      float partner = __shfl_xor(xv[e], 4, 32);
      float outv = xv[e];
      if (d >= NOPE_) {
        int r = d - NOPE_;
        float ang = rope_angle(s, r & 31);
        float c = cosf(ang), sn = sinf(ang);
        outv = (r < 32) ? (xv[e] * c + partner * sn)
                        : (xv[e] * c - partner * sn);
      }
      y[e] = outv;
      ss += outv * outv;
    }
#pragma unroll
    for (int m = 1; m < 32; m <<= 1) ss += __shfl_xor(ss, m, 32);
    float inv = rsqrtf(ss * (1.0f / D_) + EPS_);
    float gv = gate[(size_t)t * GD_ + hh];
#pragma unroll
    for (int e = 0; e < 8; ++e) {
      int d = lane * 8 + e;
      att[(size_t)t * QD_ + hh * D_ + d] = (_Float16)(y[e] * inv * pnw[d] * gv);
    }
  }
}

// ---------------------------------------------------------------------------
// Launch
// ---------------------------------------------------------------------------
extern "C" void kernel_launch(void* const* d_in, const int* in_sizes, int n_in,
                              void* d_out, int out_size, void* d_ws,
                              size_t ws_size, hipStream_t stream) {
  (void)in_sizes; (void)n_in; (void)out_size; (void)ws_size;
  const float* hidden  = (const float*)d_in[0];
  const float* v_first = (const float*)d_in[1];
  const float* W_qkv   = (const float*)d_in[2];
  const float* W_o     = (const float*)d_in[3];
  const float* qnw     = (const float*)d_in[4];
  const float* knw     = (const float*)d_in[5];
  const float* pnw     = (const float*)d_in[6];
  const float* vlam    = (const float*)d_in[7];

  char* ws = (char*)d_ws;
  size_t off = 0;
  auto alloc = [&](size_t bytes) -> void* {
    void* p = ws + off;
    off += (bytes + 255) & ~(size_t)255;
    return p;
  };
  _Float16* hid16  = (_Float16*)alloc((size_t)BS_ * HID_ * 2);
  _Float16* wqkv16 = (_Float16*)alloc((size_t)QKV_NPAD_ * HID_ * 2);  // padded rows
  _Float16* wo16   = (_Float16*)alloc((size_t)HID_ * QD_ * 2);
  float*    qkvf   = (float*)   alloc((size_t)BS_ * QKVD_ * 4);
  _Float16* qh     = (_Float16*)alloc((size_t)B_ * H_ * S_ * D_ * 2);
  _Float16* khp    = (_Float16*)alloc((size_t)B_ * KVH_ * S_ * D_ * 2);
  _Float16* vT     = (_Float16*)alloc((size_t)B_ * KVH_ * D_ * S_ * 2);
  float*    gate   = (float*)   alloc((size_t)BS_ * GD_ * 4);
  float*    Of     = (float*)   alloc((size_t)BS_ * QD_ * 4);
  _Float16* att16  = (_Float16*)alloc((size_t)BS_ * QD_ * 2);

  const int CT = 256;
  int n0 = BS_ * HID_;
  cvt_f32_f16<<<(n0 + CT - 1) / CT, CT, 0, stream>>>(hidden, hid16, n0);
  int n1 = QKVD_ * HID_;
  cvt_f32_f16<<<(n1 + CT - 1) / CT, CT, 0, stream>>>(W_qkv, wqkv16, n1);
  int n2 = HID_ * QD_;
  cvt_f32_f16<<<(n2 + CT - 1) / CT, CT, 0, stream>>>(W_o, wo16, n2);

  // qkv = hidden @ W_qkv^T : [4096,1024] x [4624,1024]^T -> [4096,4624]
  gemm_nt_f16f32<<<dim3(QKV_NBLK_, BS_ / 64), 128, 0, stream>>>(
      hid16, wqkv16, qkvf, BS_, QKVD_, HID_);

  prep_kernel<<<BS_, 256, 0, stream>>>(qkvf, v_first, qnw, knw, vlam,
                                       qh, khp, vT, gate);

  attn_kernel<<<dim3(S_ / 64, H_, B_), 128, 0, stream>>>(qh, khp, vT, Of);

  post_kernel<<<BS_, 256, 0, stream>>>(Of, gate, pnw, att16);

  // out = att @ W_o^T : [4096,4096] x [1024,4096]^T -> [4096,1024]
  gemm_nt_f16f32<<<dim3(HID_ / 128, BS_ / 64), 128, 0, stream>>>(
      att16, wo16, (float*)d_out, BS_, HID_, QD_);
}
